// Baseline_9904194584728
// MI455X (gfx1250) — matrix operand order repaired and checked by reading, not compiled, not run on hardware
//
#include <hip/hip_runtime.h>
#include <hip/hip_bf16.h>
#include <math.h>

typedef __attribute__((ext_vector_type(16))) _Float16 v16h;
typedef __attribute__((ext_vector_type(8)))  float    v8f;

#define L_TOK   1024
#define B_TOK   256
#define NTOK    (L_TOK * B_TOK)       // 262144 tokens
#define HID     64
#define SEQV    20
#define KMERV   10648                 // 22^3
#define NFRAG   32
#define KFRAG   96                    // 3*L / NFRAG

// workspace layout in floats
#define OFF_SEQW   0
#define OFF_KMERW  (OFF_SEQW + SEQV * HID)               // 1280
#define OFF_SRF    (OFF_KMERW + KMERV * HID)             // +681472
#define OFF_FRAG   (OFF_SRF + NTOK * 9)                  // +2359296
// total floats = 5,401,344  (~21.6 MB)

// ---------------------------------------------------------------------------
// Kernel 1: fuse embedding matmuls:  seqW = seq_embed@W0[0:16], kmerW = kmer_embed@W0[16:272]
// ---------------------------------------------------------------------------
__global__ void k_tables(const float* __restrict__ seq_embed,
                         const float* __restrict__ kmer_embed,
                         const float* __restrict__ W0,
                         float* __restrict__ ws)
{
    int tid = blockIdx.x * blockDim.x + threadIdx.x;
    int total = (SEQV + KMERV) * HID;
    if (tid >= total) return;
    int v = tid / HID, c = tid % HID;
    float acc = 0.f;
    if (v < SEQV) {
        const float* se = seq_embed + v * 16;
        #pragma unroll
        for (int k = 0; k < 16; ++k) acc += se[k] * W0[k * HID + c];
        ws[OFF_SEQW + v * HID + c] = acc;
    } else {
        int vv = v - SEQV;
        const float* ke = kmer_embed + (size_t)vv * 256;
        for (int k = 0; k < 256; ++k) acc += ke[k] * W0[(16 + k) * HID + c];
        ws[OFF_KMERW + vv * HID + c] = acc;
    }
}

// ---------------------------------------------------------------------------
// Kernel 2: fused MLP via WMMA.  One wave per 16-token tile; 8 waves/block,
// 8 tiles/wave, 256 blocks covers all 16384 tiles.
// ---------------------------------------------------------------------------
#define WAVES 8
#define TPW   8

__global__ __launch_bounds__(256) void k_mlp(
    const int* __restrict__ seq, const int* __restrict__ kmer,
    const float* __restrict__ pssm,
    const float* __restrict__ W0, const float* __restrict__ b0,
    const float* __restrict__ We, const float* __restrict__ be,
    const float* __restrict__ W1, const float* __restrict__ b1,
    const float* __restrict__ ws_ro, float* __restrict__ srf)
{
    __shared__ _Float16 sWe[64 * 64];       // We as f16
    __shared__ _Float16 sW0p[32 * 64];      // pssm-part of W0, K padded 21->32
    __shared__ _Float16 sW1[64 * 16];       // W1, N padded 9->16
    __shared__ _Float16 sH[WAVES][16 * 64]; // per-wave activation staging
    __shared__ int      sSeq[WAVES][16];
    __shared__ int      sKm[WAVES][16];

    const int tid  = threadIdx.x;
    const int lane = tid & 31;
    const int wid  = tid >> 5;
    const int lcol = lane & 15;   // N (or A-row M) index
    const int lhi  = lane >> 4;   // lane half

    const float* seqW  = ws_ro + OFF_SEQW;
    const float* kmerW = ws_ro + OFF_KMERW;

    // ---- stage weights to LDS (f32 -> f16) ----
    for (int i = tid; i < 64 * 64; i += 256) sWe[i] = (_Float16)We[i];
    for (int i = tid; i < 32 * 64; i += 256) {
        int k = i >> 6, c = i & 63;
        sW0p[i] = (k < 21) ? (_Float16)W0[(272 + k) * HID + c] : (_Float16)0.f;
    }
    for (int i = tid; i < 64 * 16; i += 256) {
        int k = i >> 4, n = i & 15;
        sW1[i] = (n < 9) ? (_Float16)W1[k * 9 + n] : (_Float16)0.f;
    }
    __syncthreads();

    // ---- loop-invariant B fragments (dense B: lane holds col N=lcol, halves = contiguous K) ----
    v16h bW0p[4], bWeL[4], bWeH[4], bW1L, bW1H;
    const int kb = lhi * 16;
    #pragma unroll
    for (int j = 0; j < 4; ++j) {
        v16h t0, tl, th;
        #pragma unroll
        for (int i = 0; i < 16; ++i) {
            int K = kb + i;
            t0[i] = sW0p[K * 64 + j * 16 + lcol];
            tl[i] = sWe[K * 64 + j * 16 + lcol];
            th[i] = sWe[(K + 32) * 64 + j * 16 + lcol];
        }
        bW0p[j] = t0; bWeL[j] = tl; bWeH[j] = th;
    }
    #pragma unroll
    for (int i = 0; i < 16; ++i) {
        int K = kb + i;
        bW1L[i] = sW1[K * 16 + lcol];
        bW1H[i] = sW1[(K + 32) * 16 + lcol];
    }

    // A-fragment loader: 16-bit A 16x32 layout (M = lcol; K contiguous per lane half, matching B)
    _Float16* h = sH[wid];
    auto loadA = [&](int kbase) -> v16h {
        v16h a;
        #pragma unroll
        for (int i = 0; i < 16; ++i) {
            int K = kbase + lhi * 16 + i;
            a[i] = h[lcol * 64 + K];
        }
        return a;
    };

    const int wave_global = blockIdx.x * WAVES + wid;
    for (int t = 0; t < TPW; ++t) {
        const int tile = wave_global * TPW + t;
        const int tok0 = tile * 16;

        if (lane < 16) {
            sSeq[wid][lane] = seq[tok0 + lane];
            sKm[wid][lane]  = kmer[tok0 + lane];
        }
        __syncthreads();

        // ---- h0 = seqW[seq] + kmerW[kmer] + b0, accumulated directly in C-frag layout ----
        v8f acc[4];
        #pragma unroll
        for (int j = 0; j < 4; ++j) {
            int c = j * 16 + lcol;
            float bb = b0[c];
            #pragma unroll
            for (int r = 0; r < 8; ++r) {
                int m = lhi * 8 + r;
                acc[j][r] = bb + seqW[sSeq[wid][m] * HID + c]
                               + kmerW[sKm[wid][m] * HID + c];
            }
        }
        // pssm part via WMMA (K padded 21->32)
        v16h aP;
        #pragma unroll
        for (int i = 0; i < 16; ++i) {
            int K = lhi * 16 + i;
            aP[i] = (K < 21) ? (_Float16)pssm[(size_t)(tok0 + lcol) * 21 + K]
                             : (_Float16)0.f;
        }
        #pragma unroll
        for (int j = 0; j < 4; ++j)
            acc[j] = __builtin_amdgcn_wmma_f32_16x16x32_f16(
                false, aP, false, bW0p[j], (short)0, acc[j], false, false);

        // stage h0 as f16 for A rebuild
        #pragma unroll
        for (int j = 0; j < 4; ++j)
            #pragma unroll
            for (int r = 0; r < 8; ++r)
                h[(lhi * 8 + r) * 64 + j * 16 + lcol] = (_Float16)acc[j][r];
        __syncthreads();

        // ---- two shared ReLU layers: h = relu(h @ We + be), K=64 via 2 chained WMMAs ----
        #pragma unroll
        for (int layer = 0; layer < 2; ++layer) {
            v16h aL = loadA(0);
            v16h aH = loadA(32);
            v8f accl[4];
            #pragma unroll
            for (int j = 0; j < 4; ++j) {
                int c = j * 16 + lcol;
                float bb = be[c];
                v8f a;
                #pragma unroll
                for (int r = 0; r < 8; ++r) a[r] = bb;
                a = __builtin_amdgcn_wmma_f32_16x16x32_f16(
                        false, aL, false, bWeL[j], (short)0, a, false, false);
                a = __builtin_amdgcn_wmma_f32_16x16x32_f16(
                        false, aH, false, bWeH[j], (short)0, a, false, false);
                #pragma unroll
                for (int r = 0; r < 8; ++r) a[r] = fmaxf(a[r], 0.f);
                accl[j] = a;
            }
            __syncthreads();
            #pragma unroll
            for (int j = 0; j < 4; ++j)
                #pragma unroll
                for (int r = 0; r < 8; ++r)
                    h[(lhi * 8 + r) * 64 + j * 16 + lcol] = (_Float16)accl[j][r];
            __syncthreads();
        }

        // ---- srf = h @ W1 + b1 (N padded to 16, only 9 cols stored) ----
        {
            v16h aL = loadA(0);
            v16h aH = loadA(32);
            v8f o;
            #pragma unroll
            for (int r = 0; r < 8; ++r) o[r] = (lcol < 9) ? b1[lcol] : 0.f;
            o = __builtin_amdgcn_wmma_f32_16x16x32_f16(
                    false, aL, false, bW1L, (short)0, o, false, false);
            o = __builtin_amdgcn_wmma_f32_16x16x32_f16(
                    false, aH, false, bW1H, (short)0, o, false, false);
            if (lcol < 9) {
                #pragma unroll
                for (int r = 0; r < 8; ++r) {
                    int m = lhi * 8 + r;
                    srf[(size_t)(tok0 + m) * 9 + lcol] = o[r];
                }
            }
        }
        __syncthreads();
    }
}

// ---------------------------------------------------------------------------
// pNeRF geometry helpers
// ---------------------------------------------------------------------------
struct F3 { float x, y, z; };
__device__ __forceinline__ F3 f3(float x, float y, float z) { F3 r; r.x = x; r.y = y; r.z = z; return r; }
__device__ __forceinline__ F3 fsub(F3 a, F3 b) { return f3(a.x - b.x, a.y - b.y, a.z - b.z); }
__device__ __forceinline__ F3 fcross(F3 a, F3 b) {
    return f3(a.y * b.z - a.z * b.y, a.z * b.x - a.x * b.z, a.x * b.y - a.y * b.x);
}
__device__ __forceinline__ F3 fnorm(F3 v) {
    float n = sqrtf(v.x * v.x + v.y * v.y + v.z * v.z);
    n = fmaxf(n, 1e-12f);
    float inv = 1.f / n;
    return f3(v.x * inv, v.y * inv, v.z * inv);
}
#define INIT_A0 (-0.7071067811865476f)
#define INIT_A1 ( 1.2247448713915890f)
#define INIT_B0 (-1.4142135623730951f)

// ---------------------------------------------------------------------------
// Kernel 3: fragment extension.  One thread per (fragment, batch) chain: 8192 chains x 96 steps.
// ---------------------------------------------------------------------------
__global__ void k_extend(const float* __restrict__ srf, float* __restrict__ frag)
{
    int id = blockIdx.x * blockDim.x + threadIdx.x;
    if (id >= NFRAG * B_TOK) return;
    int f = id / B_TOK, b = id % B_TOK;

    F3 A = f3(INIT_A0, INIT_A1, 0.f);
    F3 Bv = f3(INIT_B0, 0.f, 0.f);
    F3 C = f3(0.f, 0.f, 0.f);

    for (int s = 0; s < KFRAG; ++s) {
        int rr = f * KFRAG + s;            // row of c_tilde (3L, B, 3)
        int l = rr / 3, i = rr % 3;
        const float* p = srf + ((size_t)(l * B_TOK + b)) * 9 + i * 3;
        F3 ct = f3(p[0], p[1], p[2]);

        F3 bc = fnorm(fsub(C, Bv));
        F3 n  = fnorm(fcross(fsub(Bv, A), bc));
        F3 m2 = fcross(n, bc);
        F3 d  = f3(C.x + bc.x * ct.x + m2.x * ct.y + n.x * ct.z,
                   C.y + bc.y * ct.x + m2.y * ct.y + n.y * ct.z,
                   C.z + bc.z * ct.x + m2.z * ct.y + n.z * ct.z);

        float* q = frag + ((size_t)rr * B_TOK + b) * 3;
        q[0] = d.x; q[1] = d.y; q[2] = d.z;
        A = Bv; Bv = C; C = d;
    }
}

// ---------------------------------------------------------------------------
// Kernel 4: fragment assembly.  One block per batch column; 32 sequential frames,
// carry (last 3 points) kept in LDS.
// ---------------------------------------------------------------------------
__global__ __launch_bounds__(KFRAG) void k_assemble(const float* __restrict__ frag,
                                                    float* __restrict__ out)
{
    __shared__ float sc[3][3];
    int b = blockIdx.x;
    int s = threadIdx.x;   // 0..95

    if (s == 0) {
        sc[0][0] = INIT_A0; sc[0][1] = INIT_A1; sc[0][2] = 0.f;
        sc[1][0] = INIT_B0; sc[1][1] = 0.f;     sc[1][2] = 0.f;
        sc[2][0] = 0.f;     sc[2][1] = 0.f;     sc[2][2] = 0.f;
    }
    __syncthreads();

    for (int f = 0; f < NFRAG; ++f) {
        F3 A  = f3(sc[0][0], sc[0][1], sc[0][2]);
        F3 Bv = f3(sc[1][0], sc[1][1], sc[1][2]);
        F3 C  = f3(sc[2][0], sc[2][1], sc[2][2]);

        F3 bc = fnorm(fsub(C, Bv));
        F3 n  = fnorm(fcross(fsub(Bv, A), bc));
        F3 m2 = fcross(n, bc);

        int rr = f * KFRAG + s;
        const float* p = frag + ((size_t)rr * B_TOK + b) * 3;
        F3 fr = f3(p[0], p[1], p[2]);
        F3 y  = f3(C.x + bc.x * fr.x + m2.x * fr.y + n.x * fr.z,
                   C.y + bc.y * fr.x + m2.y * fr.y + n.y * fr.z,
                   C.z + bc.z * fr.x + m2.z * fr.y + n.z * fr.z);

        float* q = out + ((size_t)rr * B_TOK + b) * 3;
        q[0] = y.x; q[1] = y.y; q[2] = y.z;

        __syncthreads();
        if (s >= KFRAG - 3) {
            int r = s - (KFRAG - 3);
            sc[r][0] = y.x; sc[r][1] = y.y; sc[r][2] = y.z;
        }
        __syncthreads();
    }
}

// ---------------------------------------------------------------------------
// Launch
// ---------------------------------------------------------------------------
extern "C" void kernel_launch(void* const* d_in, const int* in_sizes, int n_in,
                              void* d_out, int out_size, void* d_ws, size_t ws_size,
                              hipStream_t stream)
{
    const int*   seq        = (const int*)d_in[0];
    const int*   kmer       = (const int*)d_in[1];
    const float* pssm       = (const float*)d_in[2];
    // d_in[3] = length (unused by the reference computation)
    const float* seq_embed  = (const float*)d_in[4];
    const float* kmer_embed = (const float*)d_in[5];
    const float* W0         = (const float*)d_in[6];
    const float* b0         = (const float*)d_in[7];
    const float* We         = (const float*)d_in[8];
    const float* be         = (const float*)d_in[9];
    const float* W1         = (const float*)d_in[10];
    const float* b1         = (const float*)d_in[11];

    float* ws   = (float*)d_ws;
    float* out  = (float*)d_out;
    float* srf  = ws + OFF_SRF;
    float* frag = ws + OFF_FRAG;

    (void)in_sizes; (void)n_in; (void)out_size; (void)ws_size;

    int total = (SEQV + KMERV) * HID;
    k_tables<<<(total + 255) / 256, 256, 0, stream>>>(seq_embed, kmer_embed, W0, ws);

    // 16384 tiles = 256 blocks * 8 waves * 8 tiles
    k_mlp<<<256, 256, 0, stream>>>(seq, kmer, pssm, W0, b0, We, be, W1, b1, ws, srf);

    k_extend<<<(NFRAG * B_TOK + 127) / 128, 128, 0, stream>>>(srf, frag);

    k_assemble<<<B_TOK, KFRAG, 0, stream>>>(frag, out);
}
